// TransformerBlock_65429531787271
// MI455X (gfx1250) — matrix operand order repaired and checked by reading, not compiled, not run on hardware
//
#include <hip/hip_runtime.h>
#include <hip/hip_bf16.h>

// ---------------------------------------------------------------------------
// TransformerBlock for MI455X (gfx1250): all GEMMs via v_wmma_f32_16x16x32_bf16
// (wave32, bf16 -> f32 accum). Attention is flash-fused (online softmax) so the
// 512MB BxHxTxT score matrix never touches HBM. Tile fills use gfx1250
// GLOBAL_LOAD_ASYNC_TO_LDS_B128 (ASYNCcnt) with double-buffered LDS stages.
// ---------------------------------------------------------------------------

#define B_  2
#define T_  2048
#define C_  1024
#define E_  1024
#define H_  16
#define HS_ 64
#define FF_ 4096
#define M_  (B_ * T_)   // 4096 rows of activations

typedef unsigned short u16;

typedef __attribute__((ext_vector_type(16))) __bf16          v16bf;
typedef __attribute__((ext_vector_type(8)))  float           v8f;
typedef __attribute__((ext_vector_type(16))) unsigned short  v16us;
typedef __attribute__((ext_vector_type(8)))  unsigned short  v8us;

// float -> bf16 with round-to-nearest-even
__device__ __forceinline__ u16 f2bf(float f) {
  unsigned u = __builtin_bit_cast(unsigned, f);
  u += 0x7FFFu + ((u >> 16) & 1u);
  return (u16)(u >> 16);
}

// Flat LDS pointer -> 32-bit LDS byte offset (LDS aperture keeps the offset in
// the low 32 bits per the ISA aperture rules).
__device__ __forceinline__ unsigned lds_off(const void* p) {
  return (unsigned)(unsigned long long)p;
}

// GLOBAL_LOAD_ASYNC_TO_LDS_B128: per-lane 16B memory->LDS DMA, ASYNCcnt-tracked.
// INST_OFFSET is added to BOTH the LDS and the memory address (ISA 10.x), so a
// single (lds,mem) register pair + offset immediates covers a multi-b128 copy.
template <int OFF>
__device__ __forceinline__ void async_ld_b128(unsigned lds_byte_off, const void* g) {
  asm volatile("global_load_async_to_lds_b128 %0, %1, off offset:%2"
               :: "v"(lds_byte_off), "v"(g), "i"(OFF) : "memory");
}
__device__ __forceinline__ void wait_async0() {
  asm volatile("s_wait_asynccnt 0" ::: "memory");
}

// A-matrix fragment (16x32 bf16, MxK), ISA 7.12.2:
// lane half h: elements e=0..7 -> K = h*8 + e ; e=8..15 -> K = h*8 + 16 + (e-8)
// => two contiguous 16B LDS chunks at element offsets (h*8) and (h*8+16).
__device__ __forceinline__ v16bf load_a_frag(const u16* base) {
  v8us lo = *(const v8us*)(base);
  v8us hi = *(const v8us*)(base + 16);
  v16us c = __builtin_shufflevector(lo, hi, 0,1,2,3,4,5,6,7,8,9,10,11,12,13,14,15);
  return __builtin_bit_cast(v16bf, c);
}
// B-matrix fragment (32x16 bf16, KxN): lane = column n, half h covers
// e=0..7 -> K = h*8 + e ; e=8..15 -> K = h*8 + 16 + (e-8) (same as A), B stored [n][k].
__device__ __forceinline__ v16bf load_b_frag(const u16* base) {
  v8us lo = *(const v8us*)(base);
  v8us hi = *(const v8us*)(base + 16);
  v16us c = __builtin_shufflevector(lo, hi, 0,1,2,3,4,5,6,7,8,9,10,11,12,13,14,15);
  return __builtin_bit_cast(v16bf, c);
}

__device__ __forceinline__ v8f wmma_bf16(v16bf a, v16bf b, v8f c) {
  return __builtin_amdgcn_wmma_f32_16x16x32_bf16(
      /*neg_a=*/false, a, /*neg_b=*/false, b,
      /*c_mod=*/(short)0, c, /*reuse_a=*/false, /*reuse_b=*/false);
}

__device__ __forceinline__ v8f v8f_zero() {
  v8f z = {0.f, 0.f, 0.f, 0.f, 0.f, 0.f, 0.f, 0.f};
  return z;
}

// ---------------------------------------------------------------------------
// f32 -> bf16 conversion kernels
// ---------------------------------------------------------------------------
__global__ void cvt_bf16_kernel(const float* __restrict__ in, u16* __restrict__ out, int n) {
  for (int i = blockIdx.x * blockDim.x + threadIdx.x; i < n; i += gridDim.x * blockDim.x)
    out[i] = f2bf(in[i]);
}

// out[c*rows + r] = bf16(in[r*cols + c])   (builds Wc^T for the 1x1 conv)
__global__ void cvt_bf16_T_kernel(const float* __restrict__ in, u16* __restrict__ out,
                                  int rows, int cols) {
  int n = rows * cols;
  for (int i = blockIdx.x * blockDim.x + threadIdx.x; i < n; i += gridDim.x * blockDim.x) {
    int r = i / cols, c = i % cols;
    out[(size_t)c * rows + r] = f2bf(in[i]);
  }
}

// ---------------------------------------------------------------------------
// Generic WMMA GEMM: Out(MxN) = A(MxK,bf16,row-major) * B(KxN,bf16,row-major) + bias
// Block = 256 thr (8 waves), tile 128x128, BK=32, double-buffered LDS stages.
// A-tile fill: async b128 copies (no VGPR roundtrip). B-tile stored transposed.
// MODE: 0 = f32 out, 1 = bf16 out, 2 = exact GELU + bf16 out.
// Requires M%128==0, N%128==0, K%32==0 (true for every call here).
// ---------------------------------------------------------------------------
template <int MODE>
__global__ __launch_bounds__(256)
void gemm_bf16_kernel(const u16* __restrict__ A, int lda,
                      const u16* __restrict__ Bm, int ldb,
                      const float* __restrict__ bias,
                      void* __restrict__ Out, int N, int K) {
  __shared__ __attribute__((aligned(64))) u16 As[2][128 * 32];   // [m][k]
  __shared__ __attribute__((aligned(64))) u16 Bs[2][128 * 32];   // [n][k] (transposed)

  const int t    = threadIdx.x;
  const int lane = t & 31;
  const int wv   = t >> 5;
  const int ln   = lane & 15;
  const int hh   = lane >> 4;
  const int wm   = (wv >> 1) * 32;   // wave row offset within 128
  const int wn   = (wv & 1) * 64;    // wave col offset within 128
  const int m0   = blockIdx.y * 128;
  const int n0   = blockIdx.x * 128;

  const int arow = t & 127;          // loader row / column index
  const int ah   = (t >> 7) * 16;    // loader k-half offset

  v8f acc[2][4];
#pragma unroll
  for (int mi = 0; mi < 2; ++mi)
#pragma unroll
    for (int ni = 0; ni < 4; ++ni) acc[mi][ni] = v8f_zero();

  auto load_tile = [&](int kb, int p) {
    // A tile: 32B per thread via two async b128 copies straight into LDS.
    const u16* asrc = A + (size_t)(m0 + arow) * lda + kb + ah;
    const unsigned adst = lds_off(&As[p][arow * 32 + ah]);
    async_ld_b128<0>(adst, asrc);
    async_ld_b128<16>(adst, asrc);
    // B tile transposed [n][k]; global reads coalesce across threads.
    const u16* bsrc = Bm + (size_t)(kb + ah) * ldb + n0 + arow;
    u16* bdst = &Bs[p][arow * 32 + ah];
#pragma unroll
    for (int kk = 0; kk < 16; ++kk)
      bdst[kk] = bsrc[(size_t)kk * ldb];
  };

  const int nk = K / 32;
  load_tile(0, 0);
  for (int k = 0; k < nk; ++k) {
    const int p = k & 1;
    wait_async0();        // async A-tile of stage p complete (this wave)
    __syncthreads();      // all waves' B ds-stores + compute on other buffer done
    if (k + 1 < nk) load_tile(k + 1, 1 - p);

    const u16* Asp = As[p];
    const u16* Bsp = Bs[p];
    v16bf af0 = load_a_frag(&Asp[(wm +      ln) * 32 + hh * 8]);
    v16bf af1 = load_a_frag(&Asp[(wm + 16 + ln) * 32 + hh * 8]);
    v16bf bfr[4];
#pragma unroll
    for (int ni = 0; ni < 4; ++ni)
      bfr[ni] = load_b_frag(&Bsp[(wn + ni * 16 + ln) * 32 + hh * 8]);
#pragma unroll
    for (int ni = 0; ni < 4; ++ni) {
      acc[0][ni] = wmma_bf16(af0, bfr[ni], acc[0][ni]);
      acc[1][ni] = wmma_bf16(af1, bfr[ni], acc[1][ni]);
    }
  }

  // Epilogue. C/D layout: VGPR r -> row r + 8*hh ; lane -> col ln.
#pragma unroll
  for (int mi = 0; mi < 2; ++mi) {
#pragma unroll
    for (int ni = 0; ni < 4; ++ni) {
      const int gn = n0 + wn + ni * 16 + ln;
      const float bv = bias ? bias[gn] : 0.f;
#pragma unroll
      for (int r = 0; r < 8; ++r) {
        const int gm = m0 + wm + mi * 16 + hh * 8 + r;
        float v = acc[mi][ni][r] + bv;
        if (MODE == 2) v = 0.5f * v * (1.0f + erff(v * 0.70710678118f));
        if (MODE == 0) ((float*)Out)[(size_t)gm * N + gn] = v;
        else           ((u16*)Out)[(size_t)gm * N + gn]   = f2bf(v);
      }
    }
  }
}

// ---------------------------------------------------------------------------
// Flash attention (non-causal, matches reference which has no mask).
// Grid: (T/128 query tiles, B*H). Block: 128 thr = 4 waves, wave = 32 q-rows.
// K/V tiles (64 keys) double-buffered in LDS; Q and K filled via async b128;
// j+2 tiles prefetched into GL2; P re-fragmented through LDS.
// ---------------------------------------------------------------------------
__global__ __launch_bounds__(128)
void flash_attn_kernel(const u16* __restrict__ Q, const u16* __restrict__ Kg_,
                       const u16* __restrict__ Vg_, u16* __restrict__ O) {
  __shared__ __attribute__((aligned(64))) u16 Qs[128 * 64];      // [qrow][dim]
  __shared__ __attribute__((aligned(64))) u16 Ks[2][64 * 64];    // [key][dim]
  __shared__ __attribute__((aligned(64))) u16 Vs[2][64 * 64];    // [dim][key] (transposed)
  __shared__ __attribute__((aligned(64))) u16 Ps[128 * 64];      // [qrow][key]

  const int t    = threadIdx.x;
  const int lane = t & 31;
  const int wv   = t >> 5;
  const int ln   = lane & 15;
  const int hh   = lane >> 4;
  const int wq0  = wv * 32;

  const int bh = blockIdx.y;
  const int bb = bh >> 4, h = bh & 15;
  const int q0 = blockIdx.x * 128;
  const size_t rowBase = (size_t)bb * T_;
  const u16* Qg = Q   + (rowBase + q0) * E_ + h * HS_;
  const u16* Kg = Kg_ +  rowBase       * E_ + h * HS_;
  const u16* Vg = Vg_ +  rowBase       * E_ + h * HS_;

  { // Q tile: one row (64 bf16 = 128B) per thread, 8 async b128 copies
    const u16* src = Qg + (size_t)t * E_;
    const unsigned qdst = lds_off(&Qs[t * 64]);
    async_ld_b128<0>(qdst, src);   async_ld_b128<16>(qdst, src);
    async_ld_b128<32>(qdst, src);  async_ld_b128<48>(qdst, src);
    async_ld_b128<64>(qdst, src);  async_ld_b128<80>(qdst, src);
    async_ld_b128<96>(qdst, src);  async_ld_b128<112>(qdst, src);
  }

  const int key = t & 63, hf = t >> 6;   // KV loader coordinates
  auto load_kv = [&](int j, int p) {
    const u16* ksrc = Kg + (size_t)(j * 64 + key) * E_ + hf * 32;
    const unsigned kdst = lds_off(&Ks[p][key * 64 + hf * 32]);
    async_ld_b128<0>(kdst, ksrc);  async_ld_b128<16>(kdst, ksrc);
    async_ld_b128<32>(kdst, ksrc); async_ld_b128<48>(kdst, ksrc);
    const u16* vsrc = Vg + (size_t)(j * 64 + key) * E_ + hf * 32;
#pragma unroll
    for (int d = 0; d < 32; ++d)
      Vs[p][(hf * 32 + d) * 64 + key] = vsrc[d];
  };

  v8f  Oacc[2][4];
  float mrow[2][8], lrow[2][8];
#pragma unroll
  for (int mi = 0; mi < 2; ++mi) {
#pragma unroll
    for (int ni = 0; ni < 4; ++ni) Oacc[mi][ni] = v8f_zero();
#pragma unroll
    for (int r = 0; r < 8; ++r) { mrow[mi][r] = -3.0e38f; lrow[mi][r] = 0.f; }
  }

  const int nj = T_ / 64;
  load_kv(0, 0);
  for (int j = 0; j < nj; ++j) {
    const int p = j & 1;
    wait_async0();        // Q (first iter) + K-tile async copies of stage p
    __syncthreads();      // V ds-stores visible; previous compute retired
    if (j + 1 < nj) load_kv(j + 1, 1 - p);
    if (j + 2 < nj) {     // GL2 prefetch two stages ahead (global_prefetch_b8)
      __builtin_prefetch(Kg + (size_t)((j + 2) * 64 + key) * E_ + hf * 32, 0, 0);
      __builtin_prefetch(Vg + (size_t)((j + 2) * 64 + key) * E_ + hf * 32, 0, 0);
    }
    const u16* Ksp = Ks[p];
    const u16* Vsp = Vs[p];

    // S = Q * K^T  (32x64 strip per wave)
    v8f S[2][4];
#pragma unroll
    for (int mi = 0; mi < 2; ++mi)
#pragma unroll
      for (int ni = 0; ni < 4; ++ni) S[mi][ni] = v8f_zero();
#pragma unroll
    for (int ks = 0; ks < 2; ++ks) {
      v16bf aq0 = load_a_frag(&Qs[(wq0 +      ln) * 64 + ks * 32 + hh * 8]);
      v16bf aq1 = load_a_frag(&Qs[(wq0 + 16 + ln) * 64 + ks * 32 + hh * 8]);
      v16bf bk[4];
#pragma unroll
      for (int ni = 0; ni < 4; ++ni)
        bk[ni] = load_b_frag(&Ksp[(ni * 16 + ln) * 64 + ks * 32 + hh * 8]);
#pragma unroll
      for (int ni = 0; ni < 4; ++ni) {
        S[0][ni] = wmma_bf16(aq0, bk[ni], S[0][ni]);
        S[1][ni] = wmma_bf16(aq1, bk[ni], S[1][ni]);
      }
    }

    // online softmax: rows live at (r + 8*hh) in each 16-lane half
#pragma unroll
    for (int mi = 0; mi < 2; ++mi) {
#pragma unroll
      for (int r = 0; r < 8; ++r) {
        float smax = -3.0e38f;
#pragma unroll
        for (int ni = 0; ni < 4; ++ni) {
          S[mi][ni][r] *= 0.125f;                 // 1/sqrt(HS)
          smax = fmaxf(smax, S[mi][ni][r]);
        }
        smax = fmaxf(smax, __shfl_xor(smax, 1, 16));
        smax = fmaxf(smax, __shfl_xor(smax, 2, 16));
        smax = fmaxf(smax, __shfl_xor(smax, 4, 16));
        smax = fmaxf(smax, __shfl_xor(smax, 8, 16));
        const float mnew  = fmaxf(mrow[mi][r], smax);
        const float alpha = __expf(mrow[mi][r] - mnew);
        mrow[mi][r] = mnew;
        float psum = 0.f;
#pragma unroll
        for (int ni = 0; ni < 4; ++ni) {
          const float pv = __expf(S[mi][ni][r] - mnew);
          S[mi][ni][r] = pv;
          psum += pv;
          Oacc[mi][ni][r] *= alpha;
        }
        psum += __shfl_xor(psum, 1, 16);
        psum += __shfl_xor(psum, 2, 16);
        psum += __shfl_xor(psum, 4, 16);
        psum += __shfl_xor(psum, 8, 16);
        lrow[mi][r] = lrow[mi][r] * alpha + psum;
        // stage P (bf16) in LDS for A-layout re-fragmentation (wave-private slice)
        const int prow = wq0 + mi * 16 + hh * 8 + r;
#pragma unroll
        for (int ni = 0; ni < 4; ++ni)
          Ps[prow * 64 + ni * 16 + ln] = f2bf(S[mi][ni][r]);
      }
    }

    // O += P * V
#pragma unroll
    for (int ks = 0; ks < 2; ++ks) {
      v16bf ap0 = load_a_frag(&Ps[(wq0 +      ln) * 64 + ks * 32 + hh * 8]);
      v16bf ap1 = load_a_frag(&Ps[(wq0 + 16 + ln) * 64 + ks * 32 + hh * 8]);
      v16bf bv[4];
#pragma unroll
      for (int ni = 0; ni < 4; ++ni)
        bv[ni] = load_b_frag(&Vsp[(ni * 16 + ln) * 64 + ks * 32 + hh * 8]);
#pragma unroll
      for (int ni = 0; ni < 4; ++ni) {
        Oacc[0][ni] = wmma_bf16(ap0, bv[ni], Oacc[0][ni]);
        Oacc[1][ni] = wmma_bf16(ap1, bv[ni], Oacc[1][ni]);
      }
    }
  }

  // normalize by row sum and store bf16
  u16* Og = O + (rowBase + q0) * E_ + h * HS_;
#pragma unroll
  for (int mi = 0; mi < 2; ++mi) {
#pragma unroll
    for (int r = 0; r < 8; ++r) {
      const float inv = 1.0f / lrow[mi][r];
      const int grow = wq0 + mi * 16 + hh * 8 + r;
#pragma unroll
      for (int ni = 0; ni < 4; ++ni)
        Og[(size_t)grow * E_ + ni * 16 + ln] = f2bf(Oacc[mi][ni][r] * inv);
    }
  }
}

// ---------------------------------------------------------------------------
// y = LayerNorm(a + b) ; writes f32 (for next residual) and bf16 (next GEMM)
// One block per row of C_=1024, 256 threads, 4 elems/thread.
// ---------------------------------------------------------------------------
__global__ __launch_bounds__(256)
void residual_ln_kernel(const float* __restrict__ A, const float* __restrict__ Bv,
                        const float* __restrict__ g, const float* __restrict__ be,
                        float* __restrict__ Yf, u16* __restrict__ Yh) {
  const int row = blockIdx.x, t = threadIdx.x;
  __shared__ float red[256];
  const float* a = A  + (size_t)row * C_;
  const float* b = Bv + (size_t)row * C_;
  float loc[4];
  float s = 0.f;
#pragma unroll
  for (int i = 0; i < 4; ++i) { loc[i] = a[t + i * 256] + b[t + i * 256]; s += loc[i]; }
  red[t] = s; __syncthreads();
  for (int st = 128; st > 0; st >>= 1) { if (t < st) red[t] += red[t + st]; __syncthreads(); }
  const float mean = red[0] * (1.0f / C_); __syncthreads();
  float vs = 0.f;
#pragma unroll
  for (int i = 0; i < 4; ++i) { const float d = loc[i] - mean; vs += d * d; }
  red[t] = vs; __syncthreads();
  for (int st = 128; st > 0; st >>= 1) { if (t < st) red[t] += red[t + st]; __syncthreads(); }
  const float rstd = rsqrtf(red[0] * (1.0f / C_) + 1e-5f); __syncthreads();
#pragma unroll
  for (int i = 0; i < 4; ++i) {
    const int c = t + i * 256;
    const float y = (loc[i] - mean) * rstd * g[c] + be[c];
    Yf[(size_t)row * C_ + c] = y;
    Yh[(size_t)row * C_ + c] = f2bf(y);
  }
}

// ---------------------------------------------------------------------------
// Host-side orchestration
// ---------------------------------------------------------------------------
extern "C" void kernel_launch(void* const* d_in, const int* in_sizes, int n_in,
                              void* d_out, int out_size, void* d_ws, size_t ws_size,
                              hipStream_t stream) {
  (void)in_sizes; (void)n_in; (void)out_size; (void)ws_size;
  const float* x    = (const float*)d_in[0];
  const float* Wq   = (const float*)d_in[1];  const float* bq = (const float*)d_in[2];
  const float* Wk   = (const float*)d_in[3];  const float* bk = (const float*)d_in[4];
  const float* Wv   = (const float*)d_in[5];  const float* bv = (const float*)d_in[6];
  const float* Wo   = (const float*)d_in[7];  const float* bo = (const float*)d_in[8];
  const float* g1   = (const float*)d_in[9];  const float* be1 = (const float*)d_in[10];
  const float* W1   = (const float*)d_in[11]; const float* b1 = (const float*)d_in[12];
  const float* W2   = (const float*)d_in[13]; const float* b2 = (const float*)d_in[14];
  const float* g2   = (const float*)d_in[15]; const float* be2 = (const float*)d_in[16];
  const float* Wc   = (const float*)d_in[17]; const float* bc = (const float*)d_in[18];
  float* out = (float*)d_out;

  // workspace bump allocator (256B aligned)
  char* ws = (char*)d_ws;
  size_t off = 0;
  auto alloc = [&](size_t bytes) -> void* {
    void* p = ws + off;
    off = (off + bytes + 255) & ~(size_t)255;
    return p;
  };
  u16* x16   = (u16*)alloc((size_t)M_ * C_ * 2);
  u16* Wq16  = (u16*)alloc((size_t)C_ * E_ * 2);
  u16* Wk16  = (u16*)alloc((size_t)C_ * E_ * 2);
  u16* Wv16  = (u16*)alloc((size_t)C_ * E_ * 2);
  u16* Wo16  = (u16*)alloc((size_t)E_ * C_ * 2);
  u16* W116  = (u16*)alloc((size_t)C_ * FF_ * 2);
  u16* W216  = (u16*)alloc((size_t)FF_ * C_ * 2);
  u16* WcT16 = (u16*)alloc((size_t)C_ * C_ * 2);
  u16* Qb    = (u16*)alloc((size_t)M_ * E_ * 2);
  u16* Kb    = (u16*)alloc((size_t)M_ * E_ * 2);
  u16* Vb    = (u16*)alloc((size_t)M_ * E_ * 2);
  u16* attn  = (u16*)alloc((size_t)M_ * E_ * 2);
  float* scratchF = (float*)alloc((size_t)M_ * C_ * 4);   // proj, then ff2
  float* y1f  = (float*)alloc((size_t)M_ * C_ * 4);
  u16*   y1h  = (u16*)alloc((size_t)M_ * C_ * 2);
  u16*   ffh  = (u16*)alloc((size_t)M_ * FF_ * 2);
  float* y2f  = (float*)alloc((size_t)M_ * C_ * 4);
  u16*   y2h  = (u16*)alloc((size_t)M_ * C_ * 2);

  // 1. conversions to bf16
  cvt_bf16_kernel<<<dim3(4096), 256, 0, stream>>>(x,  x16,  M_ * C_);
  cvt_bf16_kernel<<<dim3(2048), 256, 0, stream>>>(Wq, Wq16, C_ * E_);
  cvt_bf16_kernel<<<dim3(2048), 256, 0, stream>>>(Wk, Wk16, C_ * E_);
  cvt_bf16_kernel<<<dim3(2048), 256, 0, stream>>>(Wv, Wv16, C_ * E_);
  cvt_bf16_kernel<<<dim3(2048), 256, 0, stream>>>(Wo, Wo16, E_ * C_);
  cvt_bf16_kernel<<<dim3(8192), 256, 0, stream>>>(W1, W116, C_ * FF_);
  cvt_bf16_kernel<<<dim3(8192), 256, 0, stream>>>(W2, W216, FF_ * C_);
  cvt_bf16_T_kernel<<<dim3(2048), 256, 0, stream>>>(Wc, WcT16, C_, C_);

  // 2. QKV projections (bf16 out)
  dim3 gQKV(E_ / 128, M_ / 128);
  gemm_bf16_kernel<1><<<gQKV, 256, 0, stream>>>(x16, C_, Wq16, E_, bq, Qb, E_, C_);
  gemm_bf16_kernel<1><<<gQKV, 256, 0, stream>>>(x16, C_, Wk16, E_, bk, Kb, E_, C_);
  gemm_bf16_kernel<1><<<gQKV, 256, 0, stream>>>(x16, C_, Wv16, E_, bv, Vb, E_, C_);

  // 3. flash attention
  flash_attn_kernel<<<dim3(T_ / 128, B_ * H_), 128, 0, stream>>>(Qb, Kb, Vb, attn);

  // 4. output projection (f32) -> residual+LN1
  gemm_bf16_kernel<0><<<dim3(C_ / 128, M_ / 128), 256, 0, stream>>>(
      attn, E_, Wo16, C_, bo, scratchF, C_, E_);
  residual_ln_kernel<<<dim3(M_), 256, 0, stream>>>(x, scratchF, g1, be1, y1f, y1h);

  // 5. FFN: GELU(y1 @ W1 + b1) @ W2 + b2 -> residual+LN2
  gemm_bf16_kernel<2><<<dim3(FF_ / 128, M_ / 128), 256, 0, stream>>>(
      y1h, C_, W116, FF_, b1, ffh, FF_, C_);
  gemm_bf16_kernel<0><<<dim3(C_ / 128, M_ / 128), 256, 0, stream>>>(
      ffh, FF_, W216, C_, b2, scratchF, C_, FF_);
  residual_ln_kernel<<<dim3(M_), 256, 0, stream>>>(y1f, scratchF, g2, be2, y2f, y2h);

  // 6. 1x1 conv == y2 @ Wc^T + bc (f32 out)
  gemm_bf16_kernel<0><<<dim3(C_ / 128, M_ / 128), 256, 0, stream>>>(
      y2h, C_, WcT16, C_, bc, out, C_, C_);
}
